// MultiheadAttention_52269751992386
// MI455X (gfx1250) — matrix operand order repaired and checked by reading, not compiled, not run on hardware
//
#include <hip/hip_runtime.h>
#include <hip/hip_bf16.h>

// ---------------------------------------------------------------------------
// MHA forward for MI455X (gfx1250, wave32, WMMA bf16 w/ f32 accumulate)
// B=2, S=2048, D=1024, H=16, hd=64
// ---------------------------------------------------------------------------

typedef __attribute__((ext_vector_type(16))) __bf16 v16bf;
typedef __attribute__((ext_vector_type(8)))  float  v8f;

union FragU { uint4 u[2]; v16bf v; };

__device__ __forceinline__ unsigned short f2bf(float f) {
  unsigned int u = __float_as_uint(f);
  u += 0x7FFFu + ((u >> 16) & 1u);      // round-to-nearest-even
  return (unsigned short)(u >> 16);
}

__device__ __forceinline__ unsigned int pack2bf(float lo, float hi) {
  return (unsigned int)f2bf(lo) | ((unsigned int)f2bf(hi) << 16);
}

__device__ __forceinline__ uint4 cvt8bf(const float4 a, const float4 b) {
  uint4 r;
  r.x = pack2bf(a.x, a.y);
  r.y = pack2bf(a.z, a.w);
  r.z = pack2bf(b.x, b.y);
  r.w = pack2bf(b.z, b.w);
  return r;
}

__device__ __forceinline__ v8f wmma_bf16(const FragU& a, const FragU& b, v8f c) {
  // (neg_a, A, neg_b, B, c_mod, C, reuse_a, reuse_b)
  return __builtin_amdgcn_wmma_f32_16x16x32_bf16(false, a.v, false, b.v,
                                                 (short)0, c, false, false);
}

// ---------------------------------------------------------------------------
// Tiled GEMM:  out[m,n] = sum_k A[m,k] * W[n,k] + bias[n]
// M=4096, N=1024, K=1024.  Block tile 128x128, 8 waves, wave tile 32x64.
// Double-buffered LDS ping-pong, single barrier per K-step, global loads
// issued one full iteration ahead (latency hidden behind 8 WMMAs).
// LDS tiles are pre-swizzled into WMMA fragment layout (2x ds_load_b128/frag).
// outMode 0: bf16 -> [B,H,S,hd]   (Q, K)
// outMode 1: bf16 -> [B,H,hd,S]   (V, transposed for P*V B-fragments)
// outMode 2: f32  -> row-major [M,N] (final projection)
// ---------------------------------------------------------------------------
template <bool ABF16>
__global__ __launch_bounds__(256) void gemm_wmma(
    const void* __restrict__ Aptr,
    const float* __restrict__ W, const float* __restrict__ bias,
    float* __restrict__ outF, unsigned short* __restrict__ out16,
    int M, int N, int K, int outMode)
{
  __shared__ __align__(16) unsigned short smA[2][8 * 512];  // ping-pong A frags
  __shared__ __align__(16) unsigned short smB[2][8 * 512];  // ping-pong B frags

  const int t    = threadIdx.x;
  const int lane = t & 31;
  const int w    = t >> 5;
  const int hi   = lane >> 4;
  const int nlo  = lane & 15;
  const int m0   = blockIdx.x * 128;
  const int n0   = blockIdx.y * 128;
  const int wm   = w >> 1;   // 0..3
  const int wn   = w & 1;    // 0..1

  // staging coordinates: 2 threads per row, each covers 16 consecutive k
  const int srow = t >> 1;       // 0..127
  const int half = t & 1;        // k sub-half: 0 -> k 0..15, 1 -> k 16..31
  const int frag = srow >> 4;
  const int rIn  = srow & 15;

  const float* A32          = ABF16 ? nullptr : (const float*)Aptr;
  const unsigned short* A16 = ABF16 ? (const unsigned short*)Aptr : nullptr;

  const int KT = K >> 5;    // number of 32-wide K tiles

  // register staging (one tile in flight)
  float4 fa[4]; uint4 ua[2]; float4 fw[4];

  auto gload = [&](int kt) {
    const int koff = (kt << 5) + half * 16;
    if (ABF16) {
      const uint4* p = (const uint4*)(A16 + (size_t)(m0 + srow) * K + koff);
      ua[0] = p[0]; ua[1] = p[1];
    } else {
      const float4* p = (const float4*)(A32 + (size_t)(m0 + srow) * K + koff);
      fa[0] = p[0]; fa[1] = p[1]; fa[2] = p[2]; fa[3] = p[3];
    }
    const float4* q = (const float4*)(W + (size_t)(n0 + srow) * K + koff);
    fw[0] = q[0]; fw[1] = q[1]; fw[2] = q[2]; fw[3] = q[3];
  };

  auto sstore = [&](int buf) {
    uint4 lo8, hi8;
    if (ABF16) { lo8 = ua[0]; hi8 = ua[1]; }
    else       { lo8 = cvt8bf(fa[0], fa[1]); hi8 = cvt8bf(fa[2], fa[3]); }
    // A frag layout: k-quads 0..7 -> lane rIn, 8..15 -> lane 16+rIn, slots half*8..
    *(uint4*)&smA[buf][frag * 512 + rIn * 16 + half * 8]        = lo8;
    *(uint4*)&smA[buf][frag * 512 + (16 + rIn) * 16 + half * 8] = hi8;
    uint4 bl = cvt8bf(fw[0], fw[1]);
    uint4 bh = cvt8bf(fw[2], fw[3]);
    // B frag layout: same as A (k-quads 0..7 -> lane rIn, 8..15 -> lane 16+rIn, slots half*8..)
    const int laneB = rIn;
    *(uint4*)&smB[buf][frag * 512 + laneB * 16 + half * 8]        = bl;
    *(uint4*)&smB[buf][frag * 512 + (16 + laneB) * 16 + half * 8] = bh;
  };

  v8f acc[2][4] = {};

  gload(0);
  sstore(0);
  if (KT > 1) gload(1);

  for (int kt = 0; kt < KT; ++kt) {
    __syncthreads();   // staged buf[kt&1] visible; prior reads of buf[(kt+1)&1] done

    const unsigned short* sa = smA[kt & 1];
    const unsigned short* sb = smB[kt & 1];
    FragU a[2], b[4];
#pragma unroll
    for (int i = 0; i < 2; ++i) {
      int base = (2 * wm + i) * 512 + lane * 16;
      a[i].u[0] = *(const uint4*)&sa[base];
      a[i].u[1] = *(const uint4*)&sa[base + 8];
    }
#pragma unroll
    for (int j = 0; j < 4; ++j) {
      int base = (4 * wn + j) * 512 + lane * 16;
      b[j].u[0] = *(const uint4*)&sb[base];
      b[j].u[1] = *(const uint4*)&sb[base + 8];
    }
#pragma unroll
    for (int i = 0; i < 2; ++i)
#pragma unroll
      for (int j = 0; j < 4; ++j)
        acc[i][j] = wmma_bf16(a[i], b[j], acc[i][j]);

    if (kt + 1 < KT) {
      sstore((kt + 1) & 1);          // fill the buffer last read in iter kt-1
      if (kt + 2 < KT) gload(kt + 2);
    }
  }

  // ---- epilogue: C/D layout VGPR r -> (M = r + hi*8, N = nlo) ----
#pragma unroll
  for (int i = 0; i < 2; ++i) {
#pragma unroll
    for (int j = 0; j < 4; ++j) {
      int mbase = m0 + wm * 32 + i * 16 + hi * 8;
      int nn    = n0 + wn * 64 + j * 16 + nlo;
      float bv  = bias[nn];
#pragma unroll
      for (int r = 0; r < 8; ++r) {
        float vv = acc[i][j][r] + bv;
        int m = mbase + r;
        if (outMode == 2) {
          outF[(size_t)m * N + nn] = vv;
        } else {
          int bb = m >> 11, ss = m & 2047;     // S = 2048
          int hh = nn >> 6, dd = nn & 63;      // hd = 64
          size_t o = (outMode == 0)
            ? ((size_t)((bb * 16 + hh) * 2048 + ss) * 64 + dd)
            : ((size_t)((bb * 16 + hh) * 64 + dd) * 2048 + ss);
          out16[o] = f2bf(vv);
        }
      }
    }
  }
}

// ---------------------------------------------------------------------------
// Flash-attention: one wave per 16-row q tile, 32-wide causal k steps.
// Q,K bf16 in [B,H,S,hd]; V bf16 transposed [B,H,hd,S]; ctx bf16 -> [B,S,D].
// Working set (32MB bf16) fits in the 192MB L2 -> fragments loaded directly.
// ---------------------------------------------------------------------------
__global__ __launch_bounds__(256) void attn_wmma(
    const unsigned short* __restrict__ Q,
    const unsigned short* __restrict__ Km,
    const unsigned short* __restrict__ Vt,
    unsigned short* __restrict__ ctx)
{
  __shared__ __align__(16) unsigned short smP[8 * 512];  // per-wave 16x32 P tile

  const int t    = threadIdx.x;
  const int lane = t & 31;
  const int w    = t >> 5;
  const int hi   = lane >> 4;
  const int nlo  = lane & 15;

  const int wg  = blockIdx.x * 8 + w;   // 0..4095
  const int bh  = wg >> 7;              // 0..31  (b*16 + h)
  const int qt  = wg & 127;             // q tile within (b,h)
  const int qr0 = qt << 4;

  const unsigned short* Qb = Q  + (size_t)bh * 2048 * 64;
  const unsigned short* Kb = Km + (size_t)bh * 2048 * 64;
  const unsigned short* Vb = Vt + (size_t)bh * 64 * 2048;
  unsigned short* smPw = &smP[w * 512];

  // Q A-fragments for d=[0,32) and d=[32,64): lane row = nlo, two 8-elt chunks
  FragU qa0, qa1;
  {
    const unsigned short* qp = Qb + (size_t)(qr0 + nlo) * 64 + hi * 16;
    qa0.u[0] = *(const uint4*)(qp);
    qa0.u[1] = *(const uint4*)(qp + 8);
    qa1.u[0] = *(const uint4*)(qp + 32);
    qa1.u[1] = *(const uint4*)(qp + 40);
  }

  float m_i[8], l_i[8];
#pragma unroll
  for (int r = 0; r < 8; ++r) { m_i[r] = -1e30f; l_i[r] = 0.0f; }
  v8f ctxa[4] = {};

  const float scale = 0.125f;           // 1/sqrt(64)
  const int nsteps = (qr0 + 47) >> 5;   // ceil((qr0+16)/32)

  for (int jt = 0; jt < nsteps; ++jt) {
    const int j = jt << 5;

    // ---- scores: s0 (cols j..j+15), s1 (cols j+16..j+31), K over hd=64 ----
    const unsigned short* kp0 = Kb + (size_t)(j + nlo) * 64 + hi * 16;
    FragU kb0, kb1;
    kb0.u[0] = *(const uint4*)(kp0);
    kb0.u[1] = *(const uint4*)(kp0 + 8);
    kb1.u[0] = *(const uint4*)(kp0 + 32);
    kb1.u[1] = *(const uint4*)(kp0 + 40);
    v8f s0 = {};
    s0 = wmma_bf16(qa0, kb0, s0);
    s0 = wmma_bf16(qa1, kb1, s0);

    const unsigned short* kp1 = kp0 + 16 * 64;
    FragU kc0, kc1;
    kc0.u[0] = *(const uint4*)(kp1);
    kc0.u[1] = *(const uint4*)(kp1 + 8);
    kc1.u[0] = *(const uint4*)(kp1 + 32);
    kc1.u[1] = *(const uint4*)(kp1 + 40);
    v8f s1 = {};
    s1 = wmma_bf16(qa0, kc0, s1);
    s1 = wmma_bf16(qa1, kc1, s1);

    // ---- online softmax (row = qr0 + hi*8 + r, cols across 16 lanes) ----
    float alpha[8];
#pragma unroll
    for (int r = 0; r < 8; ++r) {
      int row = qr0 + hi * 8 + r;
      float x0 = s0[r] * scale;
      float x1 = s1[r] * scale;
      if (j + nlo > row)      x0 = -1e30f;   // causal mask
      if (j + 16 + nlo > row) x1 = -1e30f;
      float mx = fmaxf(x0, x1);
#pragma unroll
      for (int off = 1; off < 16; off <<= 1)
        mx = fmaxf(mx, __shfl_xor(mx, off, 16));
      float mnew = fmaxf(m_i[r], mx);
      float al = __expf(m_i[r] - mnew);
      float p0 = __expf(x0 - mnew);
      float p1 = __expf(x1 - mnew);
      float rs = p0 + p1;
#pragma unroll
      for (int off = 1; off < 16; off <<= 1)
        rs += __shfl_xor(rs, off, 16);
      l_i[r] = l_i[r] * al + rs;
      m_i[r] = mnew;
      alpha[r] = al;
      int prow = hi * 8 + r;
      smPw[prow * 32 + nlo]      = f2bf(p0);  // P in row-major 16x32
      smPw[prow * 32 + 16 + nlo] = f2bf(p1);
    }

#pragma unroll
    for (int dt = 0; dt < 4; ++dt)
#pragma unroll
      for (int r = 0; r < 8; ++r)
        ctxa[dt][r] *= alpha[r];

    // ---- P (16x32) as A-fragment from LDS (same-wave DS ordering) ----
    FragU pf;
    pf.u[0] = *(const uint4*)&smPw[nlo * 32 + hi * 16];
    pf.u[1] = *(const uint4*)&smPw[nlo * 32 + hi * 16 + 8];

    // ---- ctx += P * V over 4 d-tiles; V transposed so slots contiguous ----
#pragma unroll
    for (int dt = 0; dt < 4; ++dt) {
      const unsigned short* vp = Vb + (size_t)(dt * 16 + nlo) * 2048 + j + hi * 16;
      FragU vf;
      vf.u[0] = *(const uint4*)(vp);
      vf.u[1] = *(const uint4*)(vp + 8);
      ctxa[dt] = wmma_bf16(pf, vf, ctxa[dt]);
    }
  }

  // ---- normalize and write ctx bf16 as [B,S,D], D index = h*64 + d ----
  const int bb = bh >> 4, hh = bh & 15;
#pragma unroll
  for (int r = 0; r < 8; ++r) {
    float inv = 1.0f / l_i[r];
    int srow = qr0 + hi * 8 + r;
    size_t base = ((size_t)(bb * 2048 + srow)) * 1024 + hh * 64 + nlo;
#pragma unroll
    for (int dt = 0; dt < 4; ++dt)
      ctx[base + dt * 16] = f2bf(ctxa[dt][r] * inv);
  }
}

// ---------------------------------------------------------------------------
// Host launcher
// ---------------------------------------------------------------------------
extern "C" void kernel_launch(void* const* d_in, const int* in_sizes, int n_in,
                              void* d_out, int out_size, void* d_ws, size_t ws_size,
                              hipStream_t stream) {
  (void)in_sizes; (void)n_in; (void)out_size; (void)ws_size;

  const float* q_in = (const float*)d_in[0];
  const float* k_in = (const float*)d_in[1];
  const float* v_in = (const float*)d_in[2];
  // d_in[3] = mask (causal handled analytically)
  const float* Wq = (const float*)d_in[4];
  const float* bq = (const float*)d_in[5];
  const float* Wk = (const float*)d_in[6];
  const float* bk = (const float*)d_in[7];
  const float* Wv = (const float*)d_in[8];
  const float* bv = (const float*)d_in[9];
  const float* Wo = (const float*)d_in[10];
  const float* bo = (const float*)d_in[11];

  const int M = 4096, N = 1024, K = 1024;
  const size_t elems = (size_t)2 * 16 * 2048 * 64;   // 4M bf16 each
  unsigned short* qws = (unsigned short*)d_ws;
  unsigned short* kws = qws + elems;
  unsigned short* vws = kws + elems;
  unsigned short* cws = vws + elems;

  dim3 grid(M / 128, N / 128);
  gemm_wmma<false><<<grid, 256, 0, stream>>>(q_in, Wq, bq, nullptr, qws, M, N, K, 0);
  gemm_wmma<false><<<grid, 256, 0, stream>>>(k_in, Wk, bk, nullptr, kws, M, N, K, 0);
  gemm_wmma<false><<<grid, 256, 0, stream>>>(v_in, Wv, bv, nullptr, vws, M, N, K, 1);
  attn_wmma<<<dim3(512), 256, 0, stream>>>(qws, kws, vws, cws);
  gemm_wmma<true><<<grid, 256, 0, stream>>>(cws, Wo, bo, (float*)d_out, nullptr, M, N, K, 2);
}
